// CausalSelfAttention_22007412425239
// MI455X (gfx1250) — compile-verified, matrix operand order checked by reading, not run on hardware
//
#include <hip/hip_runtime.h>
#include <hip/hip_bf16.h>
#include <stdint.h>
#include <stddef.h>

// ---------------- problem constants (from reference) ----------------
#define DIMX   2048
#define HQ     16
#define HKVN   4
#define HDIM   128          // DIM / H
#define REPN   4            // HQ / HKV
#define BSZ    4
#define TLEN   2048
#define KVDIM  (HKVN * HDIM)      // 512
#define QKVN   (DIMX + 2 * KVDIM) // 3072
#define NTOK   (BSZ * TLEN)       // 8192

typedef __attribute__((ext_vector_type(16))) __bf16 v16bf;
typedef __attribute__((ext_vector_type(8)))  __bf16 v8bf;
typedef __attribute__((ext_vector_type(8)))  float  v8f;

// fp32 -> bf16 round-to-nearest-even
__device__ __forceinline__ __bf16 f2bf(float f) {
  union { float f; unsigned u; } c; c.f = f;
  unsigned r = c.u + 0x7FFFu + ((c.u >> 16) & 1u);
  unsigned short h = (unsigned short)(r >> 16);
  return __builtin_bit_cast(__bf16, h);
}

// -------------------------------------------------------------------
// elementwise fp32 -> bf16 conversion
// -------------------------------------------------------------------
__global__ void __launch_bounds__(256)
cvt_f32_bf16(const float* __restrict__ src, __bf16* __restrict__ dst, int n) {
  int i = blockIdx.x * blockDim.x + threadIdx.x;
  if (i < n) dst[i] = f2bf(src[i]);
}

// -------------------------------------------------------------------
// NT fragment loader: both A (MxK row-major) and B-as-W (NxK row-major)
// use the same per-lane packing for v_wmma_f32_16x16x32_bf16:
//   lanes 0-15 : row = lane,    K elements {k..k+7, k+16..k+23}
//   lanes16-31 : row = lane-16, K elements {k+8..k+15, k+24..k+31}
// -> two 16-byte contiguous loads per lane.
// -------------------------------------------------------------------
__device__ __forceinline__ v16bf load_frag_nt(const __bf16* __restrict__ base,
                                              int ld, int row0, int kBase) {
  const int lane = threadIdx.x & 31;
  const __bf16* p = base + (size_t)(row0 + (lane & 15)) * ld
                         + kBase + ((lane >> 4) << 3);
  v8bf lo = *(const v8bf*)p;
  v8bf hi = *(const v8bf*)(p + 16);
  v16bf f;
#pragma unroll
  for (int i = 0; i < 8; ++i) { f[i] = lo[i]; f[i + 8] = hi[i]; }
  return f;
}

// -------------------------------------------------------------------
// C[M,N] = A[M,K] * Bw[N,K]^T     (bf16 in, fp32 out)
// one wave per 64x64 C tile, K stepped by 32, 16 WMMA per step.
// -------------------------------------------------------------------
__global__ void __launch_bounds__(32)
gemm_nt_bf16_f32(const __bf16* __restrict__ A, const __bf16* __restrict__ Bw,
                 float* __restrict__ C, int M, int N, int K) {
  const int n0 = blockIdx.x * 64;
  const int m0 = blockIdx.y * 64;
  const int lane = threadIdx.x & 31;

  v8f acc[4][4] = {};
  for (int k = 0; k < K; k += 32) {
    if (k + 32 < K) {
      __builtin_prefetch(A + (size_t)(m0 + (lane & 15)) * K + k + 32, 0, 1);
      __builtin_prefetch(Bw + (size_t)(n0 + (lane & 15)) * K + k + 32, 0, 1);
    }
    v16bf af[4], bf[4];
#pragma unroll
    for (int i = 0; i < 4; ++i) af[i] = load_frag_nt(A, K, m0 + 16 * i, k);
#pragma unroll
    for (int j = 0; j < 4; ++j) bf[j] = load_frag_nt(Bw, K, n0 + 16 * j, k);
#pragma unroll
    for (int i = 0; i < 4; ++i)
#pragma unroll
      for (int j = 0; j < 4; ++j)
        acc[i][j] = __builtin_amdgcn_wmma_f32_16x16x32_bf16(
            false, af[i], false, bf[j], (short)0, acc[i][j], false, false);
  }
  // C layout: VGPR r -> row (r + 8*(lane>=16)), lane&15 -> column
  const int nl = lane & 15;
  const int mh = (lane >> 4) << 3;
#pragma unroll
  for (int i = 0; i < 4; ++i)
#pragma unroll
    for (int j = 0; j < 4; ++j) {
      float* cp = C + (size_t)(m0 + 16 * i + mh) * N + n0 + 16 * j + nl;
#pragma unroll
      for (int r = 0; r < 8; ++r) cp[(size_t)r * N] = acc[i][j][r];
    }
}

// -------------------------------------------------------------------
// fused RMSNorm + RoPE + q_gain. one wave per (token, head-slot).
// slots: 0..15 = q heads, 16..19 = k heads, 20..23 = v heads.
// qb:[B,HQ,T,HD]  kb,vb:[B,HKV,T,HD]  (bf16)
// -------------------------------------------------------------------
__global__ void __launch_bounds__(32)
normrope_kernel(const float* __restrict__ qkv, const float* __restrict__ q_gain,
                __bf16* __restrict__ qb, __bf16* __restrict__ kb,
                __bf16* __restrict__ vb) {
  const int tok  = blockIdx.x;
  const int slot = blockIdx.y;
  const int lane = threadIdx.x;
  const int b = tok / TLEN, t = tok % TLEN;

  const float* src;
  __bf16* dst;
  float gain = 1.0f;
  bool isV = false;
  if (slot < HQ) {
    src = qkv + (size_t)tok * QKVN + slot * HDIM;
    dst = qb + (((size_t)b * HQ + slot) * TLEN + t) * HDIM;
    gain = q_gain[slot];
  } else if (slot < HQ + HKVN) {
    const int g = slot - HQ;
    src = qkv + (size_t)tok * QKVN + DIMX + g * HDIM;
    dst = kb + (((size_t)b * HKVN + g) * TLEN + t) * HDIM;
  } else {
    const int g = slot - HQ - HKVN;
    src = qkv + (size_t)tok * QKVN + DIMX + KVDIM + g * HDIM;
    dst = vb + (((size_t)b * HKVN + g) * TLEN + t) * HDIM;
    isV = true;
  }

  float x0 = src[lane], x1 = src[lane + 32], x2 = src[lane + 64], x3 = src[lane + 96];
  if (isV) {
    dst[lane] = f2bf(x0); dst[lane + 32] = f2bf(x1);
    dst[lane + 64] = f2bf(x2); dst[lane + 96] = f2bf(x3);
    return;
  }
  float ss = x0 * x0 + x1 * x1 + x2 * x2 + x3 * x3;
#pragma unroll
  for (int m = 16; m >= 1; m >>= 1) ss += __shfl_xor(ss, m, 32);
  const float rn = rsqrtf(ss * (1.0f / HDIM) + 1.1920929e-7f);

  const float LOGB_64 = 9.210340371976184f / 64.0f;  // ln(10000)/64
  const float tt = (float)t;
#pragma unroll
  for (int u = 0; u < 2; ++u) {
    const int j = lane + 32 * u;          // rope pair (j, j+64)
    const float a  = (u == 0 ? x0 : x1) * rn;
    const float bb = (u == 0 ? x2 : x3) * rn;
    const float ang = tt * __expf(-(float)j * LOGB_64);
    float s, c;
    __sincosf(ang, &s, &c);
    dst[j]      = f2bf(( a * c + bb * s) * gain);
    dst[j + 64] = f2bf((-a * s + bb * c) * gain);
  }
}

// -------------------------------------------------------------------
// flash attention: one wave per (b, h, 16-row Q block).
// 32 keys per step: S = Q K^T via 8 WMMA, online softmax with 16-lane
// shfl reductions matching the C layout, P repacked through LDS to the
// A layout, V staged in LDS and gathered into B layout, O += P V (8 WMMA).
// writes y bf16 [NTOK, DIM].
// -------------------------------------------------------------------
__global__ void __launch_bounds__(32)
flash_attn_kernel(const __bf16* __restrict__ qb, const __bf16* __restrict__ kb,
                  const __bf16* __restrict__ vb, __bf16* __restrict__ y) {
  __shared__ __bf16 ldsP[16 * 32];
  __shared__ __bf16 ldsV[32 * HDIM];

  const int lane  = threadIdx.x;
  const int qblk  = blockIdx.x;   // T/16
  const int h     = blockIdx.y;   // HQ
  const int b     = blockIdx.z;   // B
  const int g     = h / REPN;
  const int qBase = qblk * 16;

  const __bf16* Q  = qb + ((size_t)b * HQ   + h) * TLEN * HDIM;
  const __bf16* Kp = kb + ((size_t)b * HKVN + g) * TLEN * HDIM;
  const __bf16* Vp = vb + ((size_t)b * HKVN + g) * TLEN * HDIM;

  v16bf qf[4];
#pragma unroll
  for (int i = 0; i < 4; ++i) qf[i] = load_frag_nt(Q, HDIM, qBase, 32 * i);

  v8f o[8] = {};
  float mrow[8], lrow[8];
#pragma unroll
  for (int r = 0; r < 8; ++r) { mrow[r] = -3.0e38f; lrow[r] = 0.0f; }

  const float scale = 0.08838834764831845f;  // 1/sqrt(128)
  const int nl = lane & 15;
  const int kh = (lane >> 4) << 3;   // K-offset half select
  const int mh = kh;                 // row half select (same expression)
  const int nsteps = (qBase + 16 + 31) / 32;

  for (int step = 0; step < nsteps; ++step) {
    const int kk = step * 32;

    // stage V tile 32 x HD into LDS (lane -> one key row)
    {
      const __bf16* vr = Vp + (size_t)(kk + lane) * HDIM;
      v8bf* dv = (v8bf*)(ldsV + lane * HDIM);
#pragma unroll
      for (int q8 = 0; q8 < 16; ++q8) dv[q8] = ((const v8bf*)vr)[q8];
    }

    // S tiles (keys kk..kk+15 and kk+16..kk+31)
    v8f s0 = {}, s1 = {};
#pragma unroll
    for (int i = 0; i < 4; ++i) {
      v16bf k0 = load_frag_nt(Kp, HDIM, kk,      32 * i);
      v16bf k1 = load_frag_nt(Kp, HDIM, kk + 16, 32 * i);
      s0 = __builtin_amdgcn_wmma_f32_16x16x32_bf16(false, qf[i], false, k0,
                                                   (short)0, s0, false, false);
      s1 = __builtin_amdgcn_wmma_f32_16x16x32_bf16(false, qf[i], false, k1,
                                                   (short)0, s1, false, false);
    }

    // scale + causal mask + online softmax update (per accumulator row)
    float corr[8];
#pragma unroll
    for (int r = 0; r < 8; ++r) {
      const int row = qBase + mh + r;
      float v0 = s0[r] * scale, v1 = s1[r] * scale;
      if (kk + nl      > row) v0 = -3.0e38f;
      if (kk + 16 + nl > row) v1 = -3.0e38f;
      float mx = fmaxf(v0, v1);
#pragma unroll
      for (int msk = 8; msk >= 1; msk >>= 1) mx = fmaxf(mx, __shfl_xor(mx, msk, 32));
      const float mnew = fmaxf(mrow[r], mx);
      corr[r] = __expf(mrow[r] - mnew);
      mrow[r] = mnew;
      const float p0 = __expf(v0 - mnew);
      const float p1 = __expf(v1 - mnew);
      s0[r] = p0; s1[r] = p1;
      float ls = p0 + p1;
#pragma unroll
      for (int msk = 8; msk >= 1; msk >>= 1) ls += __shfl_xor(ls, msk, 32);
      lrow[r] = lrow[r] * corr[r] + ls;
    }
#pragma unroll
    for (int j = 0; j < 8; ++j)
#pragma unroll
      for (int r = 0; r < 8; ++r) o[j][r] *= corr[r];

    // P (C layout) -> LDS [16 rows][32 keys] bf16
#pragma unroll
    for (int r = 0; r < 8; ++r) {
      ldsP[(mh + r) * 32 + nl]      = f2bf(s0[r]);
      ldsP[(mh + r) * 32 + 16 + nl] = f2bf(s1[r]);
    }
    asm volatile("s_wait_dscnt 0" ::: "memory");

    // P as 16x32 A fragment
    v16bf pf;
    {
      const __bf16* p = ldsP + nl * 32 + kh;
      v8bf lo = *(const v8bf*)p;
      v8bf hi = *(const v8bf*)(p + 16);
#pragma unroll
      for (int i = 0; i < 8; ++i) { pf[i] = lo[i]; pf[i + 8] = hi[i]; }
    }

    // O += P * V for each 16-wide d tile
#pragma unroll
    for (int j = 0; j < 8; ++j) {
      v16bf vf;
      const int col = j * 16 + nl;
#pragma unroll
      for (int e = 0; e < 8; ++e) {
        vf[e]     = ldsV[(kh + e)      * HDIM + col];
        vf[e + 8] = ldsV[(kh + 16 + e) * HDIM + col];
      }
      o[j] = __builtin_amdgcn_wmma_f32_16x16x32_bf16(false, pf, false, vf,
                                                     (short)0, o[j], false, false);
    }
  }

  // normalize and write y[tok, h*HD + d] as bf16
#pragma unroll
  for (int r = 0; r < 8; ++r) lrow[r] = 1.0f / lrow[r];
#pragma unroll
  for (int j = 0; j < 8; ++j)
#pragma unroll
    for (int r = 0; r < 8; ++r) {
      const int row = qBase + mh + r;
      y[((size_t)b * TLEN + row) * DIMX + h * HDIM + j * 16 + nl] =
          f2bf(o[j][r] * lrow[r]);
    }
}

// -------------------------------------------------------------------
extern "C" void kernel_launch(void* const* d_in, const int* in_sizes, int n_in,
                              void* d_out, int out_size, void* d_ws, size_t ws_size,
                              hipStream_t stream) {
  (void)in_sizes; (void)n_in; (void)out_size; (void)ws_size;
  const float* x      = (const float*)d_in[0];
  const float* Wq     = (const float*)d_in[1];
  const float* Wk     = (const float*)d_in[2];
  const float* Wv     = (const float*)d_in[3];
  const float* Wproj  = (const float*)d_in[4];
  const float* q_gain = (const float*)d_in[5];
  float* out = (float*)d_out;

  char* ws = (char*)d_ws;
  size_t off = 0;
  auto alloc = [&](size_t bytes) -> void* {
    void* p = ws + off;
    off += (bytes + 255) & ~(size_t)255;
    return p;
  };
  __bf16* xb     = (__bf16*)alloc((size_t)NTOK * DIMX * 2);
  __bf16* wqkvb  = (__bf16*)alloc((size_t)QKVN * DIMX * 2);
  __bf16* wprojb = (__bf16*)alloc((size_t)DIMX * DIMX * 2);
  float*  qkvf   = (float*) alloc((size_t)NTOK * QKVN * 4);
  __bf16* qb     = (__bf16*)alloc((size_t)BSZ * HQ   * TLEN * HDIM * 2);
  __bf16* kb     = (__bf16*)alloc((size_t)BSZ * HKVN * TLEN * HDIM * 2);
  __bf16* vb     = (__bf16*)alloc((size_t)BSZ * HKVN * TLEN * HDIM * 2);
  __bf16* yb     = (__bf16*)qkvf;  // alias: qkvf fully consumed before y written

  auto cvt = [&](const float* s, __bf16* d, int n) {
    cvt_f32_bf16<<<dim3((n + 255) / 256), dim3(256), 0, stream>>>(s, d, n);
  };
  cvt(x, xb, NTOK * DIMX);
  cvt(Wq, wqkvb, DIMX * DIMX);
  cvt(Wk, wqkvb + (size_t)DIMX * DIMX, KVDIM * DIMX);
  cvt(Wv, wqkvb + (size_t)(DIMX + KVDIM) * DIMX, KVDIM * DIMX);
  cvt(Wproj, wprojb, DIMX * DIMX);

  // QKV = x @ [Wq;Wk;Wv]^T   (fp32 out)
  gemm_nt_bf16_f32<<<dim3(QKVN / 64, NTOK / 64), dim3(32), 0, stream>>>(
      xb, wqkvb, qkvf, NTOK, QKVN, DIMX);

  // RMSNorm + RoPE + gain -> bf16 head-major buffers
  normrope_kernel<<<dim3(NTOK, HQ + 2 * HKVN), dim3(32), 0, stream>>>(
      qkvf, q_gain, qb, kb, vb);

  // flash attention -> y bf16 [NTOK, DIM]
  flash_attn_kernel<<<dim3(TLEN / 16, HQ, BSZ), dim3(32), 0, stream>>>(
      qb, kb, vb, yb);

  // out = y @ Wproj^T  (fp32)
  gemm_nt_bf16_f32<<<dim3(DIMX / 64, NTOK / 64), dim3(32), 0, stream>>>(
      yb, wprojb, out, NTOK, DIMX, DIMX);
}
